// ROMATransformerDecoder3_28467043238000
// MI455X (gfx1250) — matrix operand order repaired and checked by reading, not compiled, not run on hardware
//
#include <hip/hip_runtime.h>

typedef __attribute__((ext_vector_type(16))) __bf16 v16bf;
typedef __attribute__((ext_vector_type(8)))  float  v8f;
typedef __attribute__((ext_vector_type(4)))  unsigned int v4u;
typedef __attribute__((ext_vector_type(8)))  unsigned int v8u;
typedef unsigned short u16;

#define LAYERS 6
#define DMODEL 1024
#define DINT   256
#define NHEAD  4
#define DHEAD  64
#define DHID   512
#define BATCH  2
#define SEQ    2048
#define MTOK   (BATCH*SEQ)   /* 4096 tokens */

#define WMMA_BF(A_,B_,C_) __builtin_amdgcn_wmma_f32_16x16x32_bf16(false,(A_),false,(B_),(short)0,(C_),false,false)

__device__ __forceinline__ u16 f2bfu(float f){
  union { float f; unsigned u; } c; c.f = f;
  unsigned r = (c.u + 0x7FFFu + ((c.u >> 16) & 1u)) >> 16;   // RNE
  return (u16)r;
}
__device__ __forceinline__ __bf16 us2bf(u16 u){
  union { u16 u; __bf16 b; } c; c.u = u; return c.b;
}

// ---- CDNA5 async global->LDS copy (ASYNCcnt) ------------------------------
// dst: LDS byte offset (low 32 bits of generic shared address), 16B aligned.
__device__ __forceinline__ void async_ld_b128(unsigned lds_off, const void* gaddr){
  asm volatile("global_load_async_to_lds_b128 %0, %1, off"
               :: "v"(lds_off), "v"(gaddr) : "memory");
}
__device__ __forceinline__ void wait_async(){
  asm volatile("s_wait_asynccnt 0x0" ::: "memory");
}

// ---- CDNA5 Tensor Data Mover: 2D tile (rows x cols, 2-byte elems) -> LDS ---
// D# group0/group1 per CDNA5 ISA; 2-SGPR-group form
// (VADDR2/3 = NULL) is legal for tensors up to 2D.
__device__ __forceinline__ void tdm_load_2d(unsigned lds_off, const void* gaddr,
                                            unsigned tile0, unsigned tile1,
                                            unsigned dim0, unsigned dim1,
                                            unsigned stride0){
  unsigned long long ga = (unsigned long long)(size_t)gaddr;
  v4u g0;
  g0[0] = 1u;                                           // count=1, user desc
  g0[1] = lds_off;                                      // lds_addr
  g0[2] = (unsigned)(ga & 0xffffffffu);                 // global_addr[31:0]
  g0[3] = (unsigned)((ga >> 32) & 0x01ffffffu) | (2u << 30); // addr[56:32] | type=2
  v8u g1;
  g1[0] = 1u << 16;                                     // data_size=1 (2 bytes)
  g1[1] = (dim0 & 0xffffu) << 16;                       // tensor_dim0[15:0]
  g1[2] = ((dim0 >> 16) & 0xffffu) | ((dim1 & 0xffffu) << 16);   // dim0 hi | dim1 lo
  g1[3] = ((dim1 >> 16) & 0xffffu) | ((tile0 & 0xffffu) << 16);  // dim1 hi | tile_dim0
  g1[4] = tile1 & 0xffffu;                              // tile_dim1 (tile_dim2=0)
  g1[5] = stride0;                                      // tensor_dim0_stride[31:0]
  g1[6] = 0u;                                           // stride0 hi, stride1 lo
  g1[7] = 0u;
  asm volatile("tensor_load_to_lds %0, %1" :: "s"(g0), "s"(g1) : "memory");
}
__device__ __forceinline__ void wait_tensor(){
#if defined(__has_builtin) && __has_builtin(__builtin_amdgcn_s_wait_tensorcnt)
  __builtin_amdgcn_s_wait_tensorcnt(0);
#else
  asm volatile("s_wait_tensorcnt 0x0" ::: "memory");
#endif
}

// ---------------- elementwise helpers ----------------
__global__ __launch_bounds__(256) void k_f2bf(const float* __restrict__ in, u16* __restrict__ out, int n){
  int i = blockIdx.x*256 + threadIdx.x;
  if (i < n) out[i] = f2bfu(in[i]);
}
__global__ __launch_bounds__(256) void k_initx(const float* __restrict__ xin, float* __restrict__ xout,
                                               u16* __restrict__ xb, int n){
  int i = blockIdx.x*256 + threadIdx.x;
  if (i < n){ float v = xin[i]; xout[i] = v; xb[i] = f2bfu(v); }
}
__global__ __launch_bounds__(256) void k_resgamma(float* __restrict__ x, const float* __restrict__ hup,
                                                  const float* __restrict__ g, u16* __restrict__ xb, int n){
  int i = blockIdx.x*256 + threadIdx.x;
  if (i < n){ float v = x[i] + g[i & (DMODEL-1)] * hup[i]; x[i] = v; xb[i] = f2bfu(v); }
}

// ---------------- layernorm over DINT=256, one row per 256-thread block ----------------
__global__ __launch_bounds__(256) void k_ln(const float* __restrict__ x, const float* __restrict__ s,
                                            const float* __restrict__ b, u16* __restrict__ out){
  int row = blockIdx.x, t = threadIdx.x;
  float v = x[(size_t)row*DINT + t];
  __shared__ float sh[256];
  sh[t] = v; __syncthreads();
  for (int st = 128; st > 0; st >>= 1){ if (t < st) sh[t] += sh[t+st]; __syncthreads(); }
  float mean = sh[0] * (1.f/DINT);
  __syncthreads();
  float d = v - mean;
  sh[t] = d*d; __syncthreads();
  for (int st = 128; st > 0; st >>= 1){ if (t < st) sh[t] += sh[t+st]; __syncthreads(); }
  float r = rsqrtf(sh[0]*(1.f/DINT) + 1e-5f);
  out[(size_t)row*DINT + t] = f2bfu(d*r*s[t] + b[t]);
}

// ---------------- bf16 WMMA GEMM: C[M,N] = A[M,K] * W[N,K]^T (+bias,+gelu,+resid) ----------------
#define BM 128
#define BN 64
#define BK 32
__global__ __launch_bounds__(256) void k_gemm(const u16* __restrict__ A, const u16* __restrict__ W,
                                              const float* __restrict__ bias, const float* __restrict__ resid,
                                              float* __restrict__ outF, u16* __restrict__ outB,
                                              int M, int Nout, int K, int fuse_gelu){
  __shared__ __align__(16) u16 lA[BM*BK];
  __shared__ __align__(16) u16 lW[BN*BK];
  int tid = threadIdx.x;
  int wave = tid >> 5, lane = tid & 31, hi = lane >> 4, lo = lane & 15;
  int bm = blockIdx.x * BM, bn = blockIdx.y * BN;
  int wm = (wave >> 1) * 32, wn = (wave & 1) * 32;
  v8f acc[2][2] = {};

  for (int k0 = 0; k0 < K; k0 += BK){
    // A tile via async global->LDS (ASYNCcnt), 16B per lane-chunk
    for (int i = tid; i < (BM*BK)/8; i += 256){
      int row = (i*8) / BK, col = (i*8) % BK;
      async_ld_b128((unsigned)(size_t)&lA[i*8],
                    &A[(size_t)(bm+row)*K + k0 + col]);
    }
    // W tile via Tensor Data Mover (TENSORcnt): 2D tile BNxBK, row stride K
    if (wave == 0){
      tdm_load_2d((unsigned)(size_t)&lW[0], &W[(size_t)bn*K + k0],
                  (unsigned)BK, (unsigned)BN, (unsigned)K, (unsigned)Nout, (unsigned)K);
      wait_tensor();
    }
    if (k0 + BK < K){  // L2 prefetch of next A tile (global_prefetch_b8)
      __builtin_prefetch(&A[(size_t)(bm + (tid & (BM-1)))*K + k0 + BK], 0, 3);
    }
    wait_async();
    __syncthreads();
    v16bf a[2], b[2];
    #pragma unroll
    for (int t = 0; t < 2; ++t){
      int r = wm + t*16 + lo;
      int c = wn + t*16 + lo;
      #pragma unroll
      for (int e = 0; e < 16; ++e){
        int ka = ((e>>3)<<4) + (hi<<3) + (e&7);   // A: K split per lane-half
        a[t][e] = us2bf(lA[r*BK + ka]);
        int kb = ((e>>3)<<4) + (hi<<3) + (e&7);   // B: K split per lane-half
        b[t][e] = us2bf(lW[c*BK + kb]);
      }
    }
    #pragma unroll
    for (int i = 0; i < 2; ++i)
      #pragma unroll
      for (int j = 0; j < 2; ++j)
        acc[i][j] = WMMA_BF(a[i], b[j], acc[i][j]);
    __syncthreads();
  }
  #pragma unroll
  for (int i = 0; i < 2; ++i)
    #pragma unroll
    for (int j = 0; j < 2; ++j){
      int cb = bn + wn + j*16 + lo;
      #pragma unroll
      for (int v = 0; v < 8; ++v){
        int r = bm + wm + i*16 + v + (hi<<3);
        float val = acc[i][j][v];
        if (bias)      val += bias[cb];
        if (fuse_gelu) val = 0.5f*val*(1.f + erff(val*0.70710678f));
        if (resid)     val += resid[(size_t)r*Nout + cb];
        if (outF) outF[(size_t)r*Nout + cb] = val;
        if (outB) outB[(size_t)r*Nout + cb] = f2bfu(val);
      }
    }
}

// ---------------- qkv split + RoPE + logn + 1/sqrt(HD) folded into q ----------------
__global__ __launch_bounds__(256) void k_rope(const float* __restrict__ qkv, u16* __restrict__ qb,
                                              u16* __restrict__ kb, u16* __restrict__ vb){
  int idx = blockIdx.x*256 + threadIdx.x;        // (b,h,n,d)
  int d  = idx & (DHEAD-1);
  int n  = (idx >> 6) & (SEQ-1);
  int hh = (idx >> 17) & (NHEAD-1);
  int b  = idx >> 19;
  int s  = b*SEQ + n;
  size_t base = (size_t)s*(3*DINT) + hh*DHEAD;
  float q = qkv[base + d];
  float k = qkv[base + DINT + d];
  float v = qkv[base + 2*DINT + d];
  int j = d & 31;
  float ang = (float)n * __expf(-(float)j * (9.210340372f/32.f));  // THETA=1e4
  float c = __cosf(ang), sn = __sinf(ang);
  float qro = (d < 32) ? -qkv[base + d + 32]         : qkv[base + d - 32];
  float kro = (d < 32) ? -qkv[base + DINT + d + 32]  : qkv[base + DINT + d - 32];
  float logn = fmaxf(1.f, __logf((float)(n+1)) * (1.f/4.1588830834f)); // /ln(64)
  float qq = (q*c + qro*sn) * logn * 0.125f;   // * 1/sqrt(HD)
  float kk = k*c + kro*sn;
  size_t o = ((size_t)(b*NHEAD + hh)*SEQ + n)*DHEAD + d;
  qb[o] = f2bfu(qq); kb[o] = f2bfu(kk); vb[o] = f2bfu(v);
}

// ---------------- flash attention, WMMA for QK^T and PV ----------------
__global__ __launch_bounds__(256) void k_attn(const u16* __restrict__ qg, const u16* __restrict__ kg,
                                              const u16* __restrict__ vg, u16* __restrict__ og){
  int bh = blockIdx.x;
  int q0 = blockIdx.y * 128;
  int tid = threadIdx.x;
  int wave = tid >> 5, lane = tid & 31, hi = lane >> 4, lo = lane & 15;
  const u16* Q = qg + (size_t)bh*SEQ*DHEAD;
  const u16* K = kg + (size_t)bh*SEQ*DHEAD;
  const u16* V = vg + (size_t)bh*SEQ*DHEAD;
  __shared__ __align__(16) u16 lK[32*DHEAD];
  __shared__ __align__(16) u16 lV[32*DHEAD];
  __shared__ __align__(16) u16 lP[8][16*32];

  // Q fragments for this wave's 16 rows (A layout, K halves 0..31 / 32..63)
  v16bf qa0, qa1;
  int qr = q0 + wave*16 + lo;
  #pragma unroll
  for (int e = 0; e < 16; ++e){
    int kk = ((e>>3)<<4) + (hi<<3) + (e&7);
    qa0[e] = us2bf(Q[(size_t)qr*DHEAD + kk]);
    qa1[e] = us2bf(Q[(size_t)qr*DHEAD + 32 + kk]);
  }
  v8f o0 = {}, o1 = {}, o2 = {}, o3 = {};
  float mrow[8], lrow[8];
  #pragma unroll
  for (int v = 0; v < 8; ++v){ mrow[v] = -3.0e38f; lrow[v] = 0.f; }

  for (int j0 = 0; j0 < SEQ; j0 += 32){
    __syncthreads();
    // K/V tiles via async global->LDS copies (ASYNCcnt), 1 chunk per thread
    for (int i = tid; i < (32*DHEAD)/8; i += 256){
      async_ld_b128((unsigned)(size_t)&lK[i*8], &K[(size_t)j0*DHEAD + i*8]);
      async_ld_b128((unsigned)(size_t)&lV[i*8], &V[(size_t)j0*DHEAD + i*8]);
    }
    wait_async();
    __syncthreads();
    // S = Q * K^T for 32 keys -> two 16x16 tiles
    v16bf k0lo, k0hi, k1lo, k1hi;
    #pragma unroll
    for (int e = 0; e < 16; ++e){
      int kk = ((e>>3)<<4) + (hi<<3) + (e&7);
      k0lo[e] = us2bf(lK[lo*DHEAD + kk]);
      k0hi[e] = us2bf(lK[lo*DHEAD + 32 + kk]);
      k1lo[e] = us2bf(lK[(16+lo)*DHEAD + kk]);
      k1hi[e] = us2bf(lK[(16+lo)*DHEAD + 32 + kk]);
    }
    v8f s0 = {}, s1 = {};
    s0 = WMMA_BF(qa0, k0lo, s0); s0 = WMMA_BF(qa1, k0hi, s0);
    s1 = WMMA_BF(qa0, k1lo, s1); s1 = WMMA_BF(qa1, k1hi, s1);
    // online softmax (row = v + 8*hi, col = lo within 16-lane group)
    float alpha[8];
    #pragma unroll
    for (int v = 0; v < 8; ++v){
      float m = fmaxf(s0[v], s1[v]);
      #pragma unroll
      for (int off = 8; off; off >>= 1) m = fmaxf(m, __shfl_xor(m, off, 16));
      float mn = fmaxf(mrow[v], m);
      float p0 = __expf(s0[v]-mn), p1 = __expf(s1[v]-mn);
      float l = p0 + p1;
      #pragma unroll
      for (int off = 8; off; off >>= 1) l += __shfl_xor(l, off, 16);
      float a = __expf(mrow[v]-mn);
      alpha[v] = a;
      lrow[v] = lrow[v]*a + l;
      mrow[v] = mn;
      int row = v + hi*8;
      lP[wave][row*32 + lo]      = f2bfu(p0);
      lP[wave][row*32 + 16 + lo] = f2bfu(p1);
    }
    #pragma unroll
    for (int v = 0; v < 8; ++v){ o0[v]*=alpha[v]; o1[v]*=alpha[v]; o2[v]*=alpha[v]; o3[v]*=alpha[v]; }
    __syncthreads();
    // P (C layout) -> A fragment via LDS, then O += P * V
    v16bf pf;
    #pragma unroll
    for (int e = 0; e < 16; ++e){
      int kk = (hi<<4) + e;
      pf[e] = us2bf(lP[wave][lo*32 + kk]);
    }
    v16bf vf;
    #pragma unroll
    for (int e = 0; e < 16; ++e) vf[e] = us2bf(lV[((hi<<4)+e)*DHEAD + lo]);
    o0 = WMMA_BF(pf, vf, o0);
    #pragma unroll
    for (int e = 0; e < 16; ++e) vf[e] = us2bf(lV[((hi<<4)+e)*DHEAD + 16 + lo]);
    o1 = WMMA_BF(pf, vf, o1);
    #pragma unroll
    for (int e = 0; e < 16; ++e) vf[e] = us2bf(lV[((hi<<4)+e)*DHEAD + 32 + lo]);
    o2 = WMMA_BF(pf, vf, o2);
    #pragma unroll
    for (int e = 0; e < 16; ++e) vf[e] = us2bf(lV[((hi<<4)+e)*DHEAD + 48 + lo]);
    o3 = WMMA_BF(pf, vf, o3);
  }
  int b = bh >> 2, hh = bh & (NHEAD-1);
  #pragma unroll
  for (int v = 0; v < 8; ++v){
    float il = 1.f / lrow[v];
    int rr = q0 + wave*16 + v + hi*8;
    size_t orow = ((size_t)(b*SEQ + rr))*DINT + hh*DHEAD;
    og[orow + lo]      = f2bfu(o0[v]*il);
    og[orow + 16 + lo] = f2bfu(o1[v]*il);
    og[orow + 32 + lo] = f2bfu(o2[v]*il);
    og[orow + 48 + lo] = f2bfu(o3[v]*il);
  }
}

// ---------------- host orchestration ----------------
extern "C" void kernel_launch(void* const* d_in, const int* in_sizes, int n_in,
                              void* d_out, int out_size, void* d_ws, size_t ws_size,
                              hipStream_t stream){
  (void)in_sizes; (void)n_in; (void)out_size; (void)ws_size;
  const float* x      = (const float*)d_in[0];
  const float* down_w = (const float*)d_in[1];
  const float* down_b = (const float*)d_in[2];
  const float* nd_s   = (const float*)d_in[3];
  const float* nd_b   = (const float*)d_in[4];
  const float* qkv_w  = (const float*)d_in[5];
  const float* proj_w = (const float*)d_in[6];
  const float* proj_b = (const float*)d_in[7];
  const float* nm_s   = (const float*)d_in[8];
  const float* nm_b   = (const float*)d_in[9];
  const float* m1_w   = (const float*)d_in[10];
  const float* m1_b   = (const float*)d_in[11];
  const float* m2_w   = (const float*)d_in[12];
  const float* m2_b   = (const float*)d_in[13];
  const float* up_w   = (const float*)d_in[14];
  const float* up_b   = (const float*)d_in[15];
  const float* gamma  = (const float*)d_in[16];
  float* xout = (float*)d_out;

  char* p = (char*)d_ws;
  auto carve = [&](size_t bytes){ char* r = p; p += (bytes + 255) & ~(size_t)255; return r; };
  u16*  xb     = (u16*)carve((size_t)MTOK*DMODEL*2);
  u16*  wbdown = (u16*)carve((size_t)LAYERS*DINT*DMODEL*2);
  u16*  wbqkv  = (u16*)carve((size_t)LAYERS*3*DINT*DINT*2);
  u16*  wbproj = (u16*)carve((size_t)LAYERS*DINT*DINT*2);
  u16*  wbm1   = (u16*)carve((size_t)LAYERS*DHID*DINT*2);
  u16*  wbm2   = (u16*)carve((size_t)LAYERS*DINT*DHID*2);
  u16*  wbup   = (u16*)carve((size_t)LAYERS*DMODEL*DINT*2);
  float* h     = (float*)carve((size_t)MTOK*DINT*4);
  u16*  tb     = (u16*)carve((size_t)MTOK*DINT*2);   // reused: ln1 out / attn out / ln2 out / h-bf16
  float* qkvf  = (float*)carve((size_t)MTOK*3*DINT*4);
  u16*  qb     = (u16*)carve((size_t)BATCH*NHEAD*SEQ*DHEAD*2);
  u16*  kb     = (u16*)carve((size_t)BATCH*NHEAD*SEQ*DHEAD*2);
  u16*  vb     = (u16*)carve((size_t)BATCH*NHEAD*SEQ*DHEAD*2);
  u16*  m1q    = (u16*)carve((size_t)MTOK*DHID*2);
  float* hup   = (float*)carve((size_t)MTOK*DMODEL*4);

  auto cvt = [&](const float* src, u16* dst, size_t n){
    k_f2bf<<<dim3((unsigned)((n + 255)/256)), dim3(256), 0, stream>>>(src, dst, (int)n);
  };
  cvt(down_w, wbdown, (size_t)LAYERS*DINT*DMODEL);
  cvt(qkv_w,  wbqkv,  (size_t)LAYERS*3*DINT*DINT);
  cvt(proj_w, wbproj, (size_t)LAYERS*DINT*DINT);
  cvt(m1_w,   wbm1,   (size_t)LAYERS*DHID*DINT);
  cvt(m2_w,   wbm2,   (size_t)LAYERS*DINT*DHID);
  cvt(up_w,   wbup,   (size_t)LAYERS*DMODEL*DINT);

  k_initx<<<dim3(MTOK*DMODEL/256), dim3(256), 0, stream>>>(x, xout, xb, MTOK*DMODEL);

  for (int l = 0; l < LAYERS; ++l){
    // h = x @ down_w^T + down_b
    k_gemm<<<dim3(MTOK/BM, DINT/BN), dim3(256), 0, stream>>>(
        xb, wbdown + (size_t)l*DINT*DMODEL, down_b + l*DINT, nullptr, h, nullptr, MTOK, DINT, DMODEL, 0);
    // hn = LN(h)
    k_ln<<<dim3(MTOK), dim3(256), 0, stream>>>(h, nd_s + l*DINT, nd_b + l*DINT, tb);
    // qkv = hn @ qkv_w^T
    k_gemm<<<dim3(MTOK/BM, (3*DINT)/BN), dim3(256), 0, stream>>>(
        tb, wbqkv + (size_t)l*3*DINT*DINT, nullptr, nullptr, qkvf, nullptr, MTOK, 3*DINT, DINT, 0);
    // split + RoPE + logn + scale
    k_rope<<<dim3(BATCH*NHEAD*SEQ*DHEAD/256), dim3(256), 0, stream>>>(qkvf, qb, kb, vb);
    // o = softmax(qk^T)v  (flash, WMMA)
    k_attn<<<dim3(BATCH*NHEAD, SEQ/128), dim3(256), 0, stream>>>(qb, kb, vb, tb);
    // h = h + o @ proj_w^T + proj_b
    k_gemm<<<dim3(MTOK/BM, DINT/BN), dim3(256), 0, stream>>>(
        tb, wbproj + (size_t)l*DINT*DINT, proj_b + l*DINT, h, h, nullptr, MTOK, DINT, DINT, 0);
    // m = LN(h)
    k_ln<<<dim3(MTOK), dim3(256), 0, stream>>>(h, nm_s + l*DINT, nm_b + l*DINT, tb);
    // m1 = gelu(m @ m1w^T + m1b)  (bf16 only)
    k_gemm<<<dim3(MTOK/BM, DHID/BN), dim3(256), 0, stream>>>(
        tb, wbm1 + (size_t)l*DHID*DINT, m1_b + l*DHID, nullptr, nullptr, m1q, MTOK, DHID, DINT, 1);
    // h = h + m1 @ m2w^T + m2b   (also emit h in bf16 for 'up')
    k_gemm<<<dim3(MTOK/BM, DINT/BN), dim3(256), 0, stream>>>(
        m1q, wbm2 + (size_t)l*DINT*DHID, m2_b + l*DINT, h, h, tb, MTOK, DINT, DHID, 0);
    // hup = h @ up_w^T + up_b
    k_gemm<<<dim3(MTOK/BM, DMODEL/BN), dim3(256), 0, stream>>>(
        tb, wbup + (size_t)l*DMODEL*DINT, up_b + l*DMODEL, nullptr, hup, nullptr, MTOK, DMODEL, DINT, 0);
    // x = x + gamma * hup ; xb = bf16(x)
    k_resgamma<<<dim3(MTOK*DMODEL/256), dim3(256), 0, stream>>>(xout, hup, gamma + l*DMODEL, xb, MTOK*DMODEL);
  }
}
